// AdjacencyLSTM_635655160166
// MI455X (gfx1250) — compile-verified, matrix operand order checked by reading, not run on hardware
//
#include <hip/hip_runtime.h>
#include <cstdint>
#include <cstddef>

// ---------------------------------------------------------------------------
// AdjacencyLSTM loss for MI455X (gfx1250, wave32, WMMA 16x16x32 f16->f32).
//
// Pipeline (all on `stream`):
//   0. zero d_out (2 f32 accumulators)
//   1. pack all GEMM "B" weights to f16 in the ISA 16-bit B-operand lane layout
//   2. build feat[65024,320] (edge-exist emb | weight-MLP | pos-enc | node emb) f16
//   3. xw0 = feat @ W_ih0^T        (WMMA GEMM, M=65024 N=1024 K=320) f32
//   4. layer-0 LSTM scan (persistent 32-wave WG, h/c/gates in LDS, WMMA recurrence)
//   5. xw1 = out0 @ W_ih1^T        (WMMA GEMM, K=256)
//   6. layer-1 LSTM scan
//   7. fused 3-head MLP + BCE / Gaussian-NLL loss, LDS + global atomics
// ---------------------------------------------------------------------------

typedef __attribute__((ext_vector_type(16))) _Float16 v16h;
typedef __attribute__((ext_vector_type(8)))  float    v8f;

#define SEQ_L 8128
#define NROWS (8 * SEQ_L)   /* 65024 = B*L, b-major rows (row = b*L + t) */

__device__ __forceinline__ float sigm(float x) { return 1.0f / (1.0f + expf(-x)); }

// Load a 16x32 f16 A-fragment from LDS (stored as row-major halves, viewed as
// dwords). Per ISA: lane l holds row m=l&15; kbase = (l>>4)*8 halves; VGPR v
// holds k = kbase + (v<4 ? 2v : 16+2(v-4)) (+1 for the high half) -> dword
// index within the 16-dword tile: (l>>4)*4 + (v<4 ? v : 4+v).
__device__ __forceinline__ v16h load_a_frag(const uint32_t* s, int lane,
                                            int rowStrideDw, int dwBase) {
  union { v16h v; uint32_t u[8]; } a;
  const int m  = lane & 15;
  const int b4 = (lane >> 4) * 4;
  const uint32_t* row = s + m * rowStrideDw + dwBase;
#pragma unroll
  for (int v = 0; v < 8; ++v) {
    int d = (v < 4) ? (b4 + v) : (b4 + 4 + v);
    a.u[v] = row[d];
  }
  return a.v;
}

// ---------------------------------------------------------------------------
__global__ void zero_out_k(float* o) {
  if (threadIdx.x < 2) o[threadIdx.x] = 0.0f;
}

// Pack row-major f32 W[N,K] (gemm computes X@W^T, so B[k][n] = W[n][k]) into
// f16 tiles matching the B-operand lane layout; packed element index is
// (((nt*KT + kt)*32 + lane)*16 + j).
__global__ void pack_b16_k(const float* __restrict__ W, _Float16* __restrict__ dst,
                           int N, int K) {
  int idx = blockIdx.x * blockDim.x + threadIdx.x;
  if (idx >= N * K) return;
  int j    = idx & 15;
  int l    = (idx >> 4) & 31;
  int tile = idx >> 9;
  int KT   = K >> 5;
  int kt   = tile % KT;
  int nt   = tile / KT;
  int n    = nt * 16 + (l & 15);
  int kb   = (l >> 4) * 8;
  int v    = j >> 1;
  int kk   = ((v < 4) ? (2 * v) : (16 + 2 * (v - 4))) + (j & 1);
  int k    = kt * 32 + kb + kk;
  dst[idx] = (_Float16)W[(size_t)n * K + k];
}

// ---------------------------------------------------------------------------
// feat row = [ee(128) | we(64) | pe(64) | ne(64)], teacher-forced (prev edge).
__global__ void build_feat_k(const float* __restrict__ xadj,
                             const float* __restrict__ eee,    // [2,128]
                             const float* __restrict__ nemb,   // [128,64]
                             const float* __restrict__ ewW1, const float* __restrict__ ewb1,
                             const float* __restrict__ ewW2, const float* __restrict__ ewb2,
                             _Float16* __restrict__ featH) {
  int row = blockIdx.x * blockDim.x + threadIdx.x;
  if (row >= NROWS) return;
  int t = row % SEQ_L;                       // row = b*L + t
  float pw = 0.0f, pt = 0.0f;
  if (t > 0) {
    float xp = xadj[row - 1];
    if (xp > 0.0f) { pt = 1.0f; pw = xp; }
  }
  int ee_row = (pt > 0.0f) ? 1 : 0;
  // node id: r(r-1)/2 <= t < r(r+1)/2, r in [1,127]
  int r = (int)floorf((1.0f + sqrtf(1.0f + 8.0f * (float)t)) * 0.5f);
  while ((r * (r - 1)) / 2 > t)  --r;
  while ((r * (r + 1)) / 2 <= t) ++r;

  _Float16* frow = featH + (size_t)row * 320;
  const float ln1e4 = logf(10000.0f);
  for (int jj = 0; jj < 320; ++jj) {
    float val;
    if (jj < 128) {
      val = eee[ee_row * 128 + jj];
    } else if (jj < 192) {                   // weight-embed MLP: 1 -> 128 relu -> 64
      int j = jj - 128;
      float s = ewb2[j];
      for (int h = 0; h < 128; ++h) {
        float rv = pw * ewW1[h] + ewb1[h];
        rv = rv > 0.0f ? rv : 0.0f;
        s += ewW2[j * 128 + h] * rv;
      }
      val = s;
    } else if (jj < 256) {                   // sinusoidal pos encoding, d=64
      int idx = jj - 192;
      int i2  = idx >> 1;
      float ang = (float)t * expf(-ln1e4 * (float)(2 * i2) / 64.0f);
      val = (idx & 1) ? cosf(ang) : sinf(ang);
    } else {
      val = nemb[(size_t)r * 64 + (jj - 256)];
    }
    frow[jj] = (_Float16)val;
  }
}

// ---------------------------------------------------------------------------
// C[M,1024] = A[M, KT*32] (f16 row-major) @ B^T via packed f16 B tiles.
// Block = 256 threads (8 waves); one 16-row M-tile per block; wave w owns
// N-tiles w*8..w*8+7. A tile rows staged once into LDS, then pure WMMA.
__global__ void __launch_bounds__(256)
gemm_xw_k(const _Float16* __restrict__ A, const _Float16* __restrict__ Bp,
          float* __restrict__ C, int KT) {
  __shared__ uint32_t As[16 * 160];          // up to K=320 halves/row
  const int KT16 = KT * 16;                  // dwords per staged row
  const int lda  = KT * 32;                  // halves per A row
  const int mt   = blockIdx.x;
  const int tid  = threadIdx.x;
  const int wave = tid >> 5, lane = tid & 31;

  for (int i = tid; i < 16 * KT16; i += 256) {
    int row = i / KT16, dw = i - row * KT16;
    As[row * KT16 + dw] =
        *(const uint32_t*)(A + ((size_t)(mt * 16 + row)) * lda + dw * 2);
  }
  __syncthreads();

  v8f acc[8];
#pragma unroll
  for (int i = 0; i < 8; ++i)
#pragma unroll
    for (int rr = 0; rr < 8; ++rr) acc[i][rr] = 0.0f;

  for (int kt = 0; kt < KT; ++kt) {
    v16h a = load_a_frag(As, lane, KT16, kt * 16);
#pragma unroll
    for (int i = 0; i < 8; ++i) {
      int nt = wave * 8 + i;
      const _Float16* bp = Bp + (((size_t)nt * KT + kt) * 32 + lane) * 16;
      __builtin_prefetch(bp + 512, 0, 1);    // next-kt B tile -> global_prefetch
      v16h b = *(const v16h*)bp;
      acc[i] = __builtin_amdgcn_wmma_f32_16x16x32_f16(
          false, a, false, b, (short)0, acc[i], false, false);
    }
  }
#pragma unroll
  for (int i = 0; i < 8; ++i) {
    int nt = wave * 8 + i;
    int n  = nt * 16 + (lane & 15);
#pragma unroll
    for (int rr = 0; rr < 8; ++rr) {
      int m = rr + 8 * (lane >> 4);
      C[((size_t)(mt * 16 + m)) * 1024 + n] = acc[i][rr];
    }
  }
}

// ---------------------------------------------------------------------------
// Persistent single-workgroup LSTM scan (one layer). 1024 threads = 32 waves.
// State in LDS: hH f16 [16][256] (batch padded 8->16, rows 8..15 stay zero),
// c f32 [8][256], gates f32 [8][1024]. Per step: gates = xw[t] + h @ Whh^T
// (WMMA, wave w owns gate N-tiles 2w,2w+1), then elementwise i,f,g,o update.
__global__ void __launch_bounds__(1024)
lstm_scan_k(const float* __restrict__ xw, const _Float16* __restrict__ WhhP,
            _Float16* __restrict__ outH,
            const float* __restrict__ ih, const float* __restrict__ ic) {
  __shared__ uint32_t hHu[16 * 128];         // 16 rows x 256 halves
  __shared__ float    cS[8 * 256];
  __shared__ float    gS[8 * 1024];
  const int tid = threadIdx.x, wave = tid >> 5, lane = tid & 31;
  _Float16* hHh = (_Float16*)hHu;

  for (int i = tid; i < 16 * 256; i += 1024) {
    int m = i >> 8, j = i & 255;
    hHh[i] = (m < 8) ? (_Float16)ih[j & 127] : (_Float16)0.0f;   // tile(init_h,2)
  }
  for (int i = tid; i < 8 * 256; i += 1024) cS[i] = ic[i & 127];
  __syncthreads();

  for (int t = 0; t < SEQ_L; ++t) {
    v8f acc[2];
#pragma unroll
    for (int q = 0; q < 2; ++q) {
      int nt = wave * 2 + q;
#pragma unroll
      for (int rr = 0; rr < 8; ++rr) {
        int m = rr + 8 * (lane >> 4);        // m = batch row (8..15 padding)
        acc[q][rr] = (m < 8)
            ? xw[((size_t)m * SEQ_L + t) * 1024 + nt * 16 + (lane & 15)]
            : 0.0f;
      }
    }
#pragma unroll
    for (int kt = 0; kt < 8; ++kt) {
      v16h a = load_a_frag(hHu, lane, 128, kt * 16);
#pragma unroll
      for (int q = 0; q < 2; ++q) {
        int nt = wave * 2 + q;
        v16h b = *(const v16h*)(WhhP + (((size_t)nt * 8 + kt) * 32 + lane) * 16);
        acc[q] = __builtin_amdgcn_wmma_f32_16x16x32_f16(
            false, a, false, b, (short)0, acc[q], false, false);
      }
    }
    if (lane < 16) {                         // lanes 0..15 hold batch rows 0..7
#pragma unroll
      for (int q = 0; q < 2; ++q) {
        int n = (wave * 2 + q) * 16 + lane;
#pragma unroll
        for (int rr = 0; rr < 8; ++rr) gS[rr * 1024 + n] = acc[q][rr];
      }
    }
    __syncthreads();
#pragma unroll
    for (int p = 0; p < 2; ++p) {
      int idx = tid + p * 1024;              // 2048 = 8 rows x 256 units
      int m = idx >> 8, j = idx & 255;
      float gi = gS[m * 1024 + j];
      float gf = gS[m * 1024 + 256 + j];
      float gg = gS[m * 1024 + 512 + j];
      float go = gS[m * 1024 + 768 + j];
      float c  = sigm(gf) * cS[m * 256 + j] + sigm(gi) * tanhf(gg);
      float h  = sigm(go) * tanhf(c);
      cS[m * 256 + j]  = c;
      hHh[m * 256 + j] = (_Float16)h;
      outH[((size_t)m * SEQ_L + t) * 256 + j] = (_Float16)h;
    }
    __syncthreads();
  }
}

// ---------------------------------------------------------------------------
// Fused heads + loss. Per block: one 16-row tile of out1. 96 (head,nt) tile
// columns over 8 waves x 12 iters. hidden = relu(out1@W1^T + b1); scalar =
// hidden@W2 + b2 reduced via ds_add_f32; then BCE + masked Gaussian NLL,
// scaled by 1/(B*M)=1/1024, accumulated with global f32 atomics.
__global__ void __launch_bounds__(256)
heads_loss_k(const _Float16* __restrict__ out1H, const _Float16* __restrict__ headP,
             const float* __restrict__ lgb1, const float* __restrict__ mub1,
             const float* __restrict__ varb1,
             const float* __restrict__ lgW2, const float* __restrict__ muW2,
             const float* __restrict__ varW2,
             const float* __restrict__ lgb2, const float* __restrict__ mub2,
             const float* __restrict__ varb2,
             const float* __restrict__ xadj, float* __restrict__ outv) {
  __shared__ uint32_t As[16 * 128];
  __shared__ float    hsum[48];              // [head][row]
  const int tid = threadIdx.x, wave = tid >> 5, lane = tid & 31;
  const int mt = blockIdx.x;

  for (int i = tid; i < 2048; i += 256) {
    int row = i >> 7, dw = i & 127;
    As[i] = *(const uint32_t*)(out1H + ((size_t)(mt * 16 + row)) * 256 + dw * 2);
  }
  if (tid < 48) hsum[tid] = 0.0f;
  __syncthreads();

  const float* b1s[3] = {lgb1, mub1, varb1};
  const float* w2s[3] = {lgW2, muW2, varW2};
  for (int i = 0; i < 12; ++i) {
    int tc   = wave + 8 * i;                 // 0..95
    int head = tc >> 5;                      // 0=logits, 1=mu, 2=logvar
    int nt   = tc & 31;
    v8f acc;
#pragma unroll
    for (int rr = 0; rr < 8; ++rr) acc[rr] = 0.0f;
#pragma unroll
    for (int kt = 0; kt < 8; ++kt) {
      v16h a = load_a_frag(As, lane, 128, kt * 16);
      v16h b = *(const v16h*)(headP + (size_t)head * 131072 +
                              (((size_t)nt * 8 + kt) * 32 + lane) * 16);
      acc = __builtin_amdgcn_wmma_f32_16x16x32_f16(
          false, a, false, b, (short)0, acc, false, false);
    }
    int n = nt * 16 + (lane & 15);
    float b1v = b1s[head][n];
    float w2v = w2s[head][n];
#pragma unroll
    for (int rr = 0; rr < 8; ++rr) {
      float hv = acc[rr] + b1v;
      hv = hv > 0.0f ? hv : 0.0f;
      int m = rr + 8 * (lane >> 4);
      atomicAdd(&hsum[head * 16 + m], hv * w2v);   // ds_add_f32
    }
  }
  __syncthreads();
  if (tid < 16) {
    int   row = mt * 16 + tid;
    float xv  = xadj[row];
    float xt  = xv > 0.0f ? 1.0f : 0.0f;
    float logit = hsum[0 * 16 + tid] + lgb2[0];
    float mu    = hsum[1 * 16 + tid] + mub2[0];
    float lv    = hsum[2 * 16 + tid] + varb2[0];
    float bce = fmaxf(logit, 0.0f) - logit * xt + log1pf(expf(-fabsf(logit)));
    float wsafe = (xt > 0.0f) ? xv : 1.0f;
    float sm = (wsafe > 20.0f) ? wsafe : logf(expm1f(fminf(wsafe, 20.0f)));
    float lw = 0.5f * xt * (lv + (mu - sm) * (mu - sm) * expf(-lv));
    const float invn = 1.0f / 1024.0f;       // n = B*M = 8*128
    atomicAdd(&outv[0], bce * invn);
    atomicAdd(&outv[1], lw * invn);
  }
}

// ---------------------------------------------------------------------------
extern "C" void kernel_launch(void* const* d_in, const int* in_sizes, int n_in,
                              void* d_out, int out_size, void* d_ws, size_t ws_size,
                              hipStream_t stream) {
  (void)in_sizes; (void)n_in; (void)out_size; (void)ws_size;
  const float* x_adj  = (const float*)d_in[0];
  const float* ee_emb = (const float*)d_in[1];
  const float* n_emb  = (const float*)d_in[2];
  const float* ew_W1  = (const float*)d_in[3];
  const float* ew_b1  = (const float*)d_in[4];
  const float* ew_W2  = (const float*)d_in[5];
  const float* ew_b2  = (const float*)d_in[6];
  const float* W_ih0  = (const float*)d_in[7];
  const float* W_hh0  = (const float*)d_in[8];
  const float* W_ih1  = (const float*)d_in[9];
  const float* W_hh1  = (const float*)d_in[10];
  const float* mu_W1  = (const float*)d_in[11];
  const float* mu_b1  = (const float*)d_in[12];
  const float* mu_W2  = (const float*)d_in[13];
  const float* mu_b2  = (const float*)d_in[14];
  const float* var_W1 = (const float*)d_in[15];
  const float* var_b1 = (const float*)d_in[16];
  const float* var_W2 = (const float*)d_in[17];
  const float* var_b2 = (const float*)d_in[18];
  const float* lg_W1  = (const float*)d_in[19];
  const float* lg_b1  = (const float*)d_in[20];
  const float* lg_W2  = (const float*)d_in[21];
  const float* lg_b2  = (const float*)d_in[22];
  const float* init_h = (const float*)d_in[23];
  const float* init_c = (const float*)d_in[24];
  float* out = (float*)d_out;

  char* ws = (char*)d_ws;
  size_t off = 0;
  auto take = [&](size_t bytes) {
    size_t o = off; off += (bytes + 255) & ~(size_t)255; return o;
  };
  _Float16* featH = (_Float16*)(ws + take((size_t)NROWS * 320 * 2));
  _Float16* pIh0  = (_Float16*)(ws + take((size_t)1024 * 320 * 2));
  _Float16* pHh0  = (_Float16*)(ws + take((size_t)1024 * 256 * 2));
  _Float16* pIh1  = (_Float16*)(ws + take((size_t)1024 * 256 * 2));
  _Float16* pHh1  = (_Float16*)(ws + take((size_t)1024 * 256 * 2));
  _Float16* pHead = (_Float16*)(ws + take((size_t)3 * 512 * 256 * 2));
  float*    xw0   = (float*)(ws + take((size_t)NROWS * 1024 * 4));
  float*    xw1   = (float*)(ws + take((size_t)NROWS * 1024 * 4));
  _Float16* o0H   = (_Float16*)(ws + take((size_t)NROWS * 256 * 2));
  _Float16* o1H   = (_Float16*)(ws + take((size_t)NROWS * 256 * 2));

  zero_out_k<<<1, 32, 0, stream>>>(out);

  auto pack = [&](const float* W, _Float16* dst, int N, int K) {
    int total = N * K;
    pack_b16_k<<<(total + 255) / 256, 256, 0, stream>>>(W, dst, N, K);
  };
  pack(W_ih0, pIh0, 1024, 320);
  pack(W_hh0, pHh0, 1024, 256);
  pack(W_ih1, pIh1, 1024, 256);
  pack(W_hh1, pHh1, 1024, 256);
  pack(lg_W1,  pHead,              512, 256);   // head 0
  pack(mu_W1,  pHead + 131072,     512, 256);   // head 1
  pack(var_W1, pHead + 2 * 131072, 512, 256);   // head 2

  build_feat_k<<<(NROWS + 255) / 256, 256, 0, stream>>>(
      x_adj, ee_emb, n_emb, ew_W1, ew_b1, ew_W2, ew_b2, featH);

  gemm_xw_k<<<NROWS / 16, 256, 0, stream>>>(featH, pIh0, xw0, 10);
  lstm_scan_k<<<1, 1024, 0, stream>>>(xw0, pHh0, o0H, init_h, init_c);
  gemm_xw_k<<<NROWS / 16, 256, 0, stream>>>(o0H, pIh1, xw1, 8);
  lstm_scan_k<<<1, 1024, 0, stream>>>(xw1, pHh1, o1H, init_h, init_c);

  heads_loss_k<<<NROWS / 16, 256, 0, stream>>>(
      o1H, pHead, lg_b1, mu_b1, var_b1, lg_W2, mu_W2, var_W2,
      lg_b2, mu_b2, var_b2, x_adj, out);
}
